// ConfNetDSS_45492293599704
// MI455X (gfx1250) — matrix operand order repaired and checked by reading, not compiled, not run on hardware
//
#include <hip/hip_runtime.h>
#include <hip/hip_bf16.h>
#include <math.h>

typedef __attribute__((ext_vector_type(16))) _Float16 v16h;
typedef __attribute__((ext_vector_type(8)))  _Float16 v8h;
typedef __attribute__((ext_vector_type(8)))  float    v8f;

// ---------------- problem constants (match reference) ----------------
constexpr int H = 128, F = 64, G = 50, L = 4;
constexpr int M_MOL  = 256, K_CONF = 8, A_ATOM = 32;
constexpr int N_TOPO = M_MOL * A_ATOM;              // 8192
constexpr int N_NODE = M_MOL * K_CONF * A_ATOM;     // 65536
constexpr int NCONF  = M_MOL * K_CONF;              // 2048
constexpr int E_CONF = N_NODE * 16;                 // 1048576
constexpr int E_GRAPH = N_TOPO * 4;                 // 32768
constexpr int BOND_TYPES = 10;
constexpr float CUTF   = 10.0f;
constexpr float GSTEP  = CUTF / (G - 1);
constexpr float GCOEF  = -0.5f / (GSTEP * GSTEP);
constexpr float PI_F   = 3.14159265358979323846f;

// ---------------- WMMA fragment layouts (wave32, v_wmma_f32_16x16x32_f16) ---------
// A (16x32 f16): lane M = lane&15, half = lane>>4.
//   per-lane elements = two contiguous 8-half runs: K = half*8 + [0..7] and 16+half*8+[0..7]
// B (32x16 f16): lane N = lane&15, half = lane>>4.
//   per-lane elements = two contiguous 8-half runs, as for A: K = half*8 + [0..7] and 16+half*8+[0..7]
// C/D (16x16 f32): element i: row M = i + 8*half, col N = lane&15

__device__ __forceinline__ v16h frag_a_from_lds(const _Float16* base) {
    // base points at K=0 of this lane's row chunk (already includes kt*32 + half*8)
    v8h lo = *(const v8h*)(base);
    v8h hi = *(const v8h*)(base + 16);
    return __builtin_shufflevector(lo, hi, 0,1,2,3,4,5,6,7,8,9,10,11,12,13,14,15);
}
__device__ __forceinline__ v16h frag_b_from_lds(const _Float16* base) {
    // base points at this lane's (col, kt*32 + half*8) in K-major storage; runs at +0 and +16
    v8h lo = *(const v8h*)(base);
    v8h hi = *(const v8h*)(base + 16);
    return __builtin_shufflevector(lo, hi, 0,1,2,3,4,5,6,7,8,9,10,11,12,13,14,15);
}

// ---------------- float atomic max via int/uint trick ----------------
__device__ __forceinline__ void atomicMaxF(float* addr, float val) {
    if (val >= 0.0f) atomicMax(reinterpret_cast<int*>(addr), __float_as_int(val));
    else             atomicMin(reinterpret_cast<unsigned int*>(addr), __float_as_uint(val));
}

// ---------------- utility kernels ----------------
__global__ void fill_kernel(float* __restrict__ p, float v, size_t n) {
    size_t i = (size_t)blockIdx.x * blockDim.x + threadIdx.x;
    if (i < n) p[i] = v;
}

__global__ void edge_kernel(const float* __restrict__ pos,
                            const int* __restrict__ erow, const int* __restrict__ ecol,
                            float* __restrict__ ew, float* __restrict__ cwin) {
    int e = blockIdx.x * blockDim.x + threadIdx.x;
    if (e >= E_CONF) return;
    int r = erow[e], c = ecol[e];
    float dx = pos[3 * r + 0] - pos[3 * c + 0];
    float dy = pos[3 * r + 1] - pos[3 * c + 1];
    float dz = pos[3 * r + 2] - pos[3 * c + 2];
    float d = sqrtf(dx * dx + dy * dy + dz * dz + 1e-12f);
    ew[e] = d;
    cwin[e] = 0.5f * (cosf(d * (PI_F / CUTF)) + 1.0f);
}

__global__ void embed_kernel(const float* __restrict__ emb, const int* __restrict__ x_topo,
                             const int* __restrict__ cnb, float* __restrict__ x) {
    size_t idx = (size_t)blockIdx.x * blockDim.x + threadIdx.x;
    if (idx >= (size_t)N_NODE * H) return;
    int n = (int)(idx >> 7), hh = (int)(idx & 127);
    x[idx] = emb[(size_t)x_topo[cnb[n]] * H + hh];
}

// scatter segment-max: src [rows,H] -> dst[seg[n],H]
__global__ void smax_kernel(const float* __restrict__ src, const int* __restrict__ seg,
                            float* __restrict__ dst, int rows) {
    size_t idx = (size_t)blockIdx.x * blockDim.x + threadIdx.x;
    if (idx >= (size_t)rows * H) return;
    int n = (int)(idx >> 7), hh = (int)(idx & 127);
    atomicMaxF(&dst[(size_t)seg[n] * H + hh], src[idx]);
}

__global__ void gin_msg_kernel(const float* __restrict__ xagg, const float* __restrict__ bemb,
                               const int* __restrict__ gr, const int* __restrict__ gc,
                               const int* __restrict__ eattr, float* __restrict__ magg) {
    size_t idx = (size_t)blockIdx.x * blockDim.x + threadIdx.x;
    if (idx >= (size_t)E_GRAPH * H) return;
    int e = (int)(idx >> 7), hh = (int)(idx & 127);
    float v = xagg[(size_t)gr[e] * H + hh] + bemb[(size_t)eattr[e] * H + hh];
    v = fmaxf(v, 0.0f);
    atomicAdd(&magg[(size_t)gc[e] * H + hh], v);
}

__global__ void gin_pre_kernel(const float* __restrict__ xagg, float* __restrict__ magg,
                               const float* __restrict__ gin_eps, int l) {
    size_t idx = (size_t)blockIdx.x * blockDim.x + threadIdx.x;
    if (idx >= (size_t)N_TOPO * H) return;
    magg[idx] += (1.0f + gin_eps[l]) * xagg[idx];
}

__global__ void combine_kernel(const float* __restrict__ cfout, const float* __restrict__ hg,
                               const int* __restrict__ cnb, float* __restrict__ x) {
    size_t idx = (size_t)blockIdx.x * blockDim.x + threadIdx.x;
    if (idx >= (size_t)N_NODE * H) return;
    int n = (int)(idx >> 7), hh = (int)(idx & 127);
    x[idx] = fmaxf(cfout[idx] + hg[(size_t)cnb[n] * H + hh], 0.0f);
}

// ---------------- batchnorm (training-mode batch stats) ----------------
__global__ void bn_stats_kernel(const float* __restrict__ y, int rows,
                                const float* __restrict__ g, const float* __restrict__ b,
                                float* __restrict__ scale, float* __restrict__ shift) {
    __shared__ float ssum[256], ssq[256];
    int ch = blockIdx.x, tid = threadIdx.x;
    float s = 0.0f, q = 0.0f;
    for (int r = tid; r < rows; r += 256) {
        float v = y[(size_t)r * H + ch];
        s += v; q += v * v;
    }
    ssum[tid] = s; ssq[tid] = q;
    __syncthreads();
    for (int o = 128; o > 0; o >>= 1) {
        if (tid < o) { ssum[tid] += ssum[tid + o]; ssq[tid] += ssq[tid + o]; }
        __syncthreads();
    }
    if (tid == 0) {
        float mu  = ssum[0] / rows;
        float var = ssq[0] / rows - mu * mu;
        float rs  = rsqrtf(var + 1e-5f);
        float sc  = g[ch] * rs;
        scale[ch] = sc;
        shift[ch] = b[ch] - mu * sc;
    }
}

__global__ void bn_apply_kernel(float* __restrict__ y, const float* __restrict__ sc,
                                const float* __restrict__ sh, int do_relu) {
    size_t idx = (size_t)blockIdx.x * blockDim.x + threadIdx.x;
    if (idx >= (size_t)N_TOPO * H) return;
    int ch = (int)(idx & 127);
    float v = y[idx] * sc[ch] + sh[ch];
    if (do_relu) v = fmaxf(v, 0.0f);
    y[idx] = v;
}

// ---------------- generic WMMA GEMM: C[M,ND] = op(A[M,KD] @ W[KD,ND] (+bias) (+X)) ----
// 4 waves / block, one 16-row tile per wave. W staged once per block into LDS,
// transposed to N-major with +8-half row padding so every B fragment is two
// conflict-free ds_load_b128s. A tiles use the same padded layout.
template<int KD, int ND, bool BIAS, bool RELU, bool ADDX>
__global__ void __launch_bounds__(128)
gemm_kernel(const float* __restrict__ A, const float* __restrict__ W,
            const float* __restrict__ bias, const float* __restrict__ X,
            float* __restrict__ C) {
    constexpr int KT = KD / 32, NT = ND / 16;
    constexpr int KDP = KD + 8;                 // padded K pitch (halves)
    __shared__ __align__(16) _Float16 sW[ND * KDP];
    __shared__ __align__(16) _Float16 sA[4][16 * KDP];
    const int tid = threadIdx.x;
    const int wv = tid >> 5, lane = tid & 31;
    const int half = lane >> 4, nn = lane & 15;
    const long row0 = ((long)blockIdx.x * 4 + wv) * 16;

    // stage W (K-major per output column), coalesced global reads
    for (int idx = tid; idx < KD * ND; idx += 128) {
        int k = idx / ND, n = idx - k * ND;
        sW[n * KDP + k] = (_Float16)W[idx];
    }
    // stage this wave's 16xKD A tile (padded rows)
    _Float16* tile = sA[wv];
    for (int idx = lane; idx < 16 * KD; idx += 32) {
        int r = idx / KD, c = idx - r * KD;
        tile[r * KDP + c] = (_Float16)A[(row0 + r) * (long)KD + c];
    }
    __syncthreads();

    v8f acc[NT] = {};
    #pragma unroll
    for (int kt = 0; kt < KT; ++kt) {
        v16h a = frag_a_from_lds(tile + nn * KDP + kt * 32 + half * 8);
        #pragma unroll
        for (int nt = 0; nt < NT; ++nt) {
            v16h b = frag_b_from_lds(sW + (nt * 16 + nn) * KDP + kt * 32 + half * 8);
            acc[nt] = __builtin_amdgcn_wmma_f32_16x16x32_f16(
                false, a, false, b, (short)0, acc[nt], false, false);
        }
    }
    #pragma unroll
    for (int nt = 0; nt < NT; ++nt) {
        int col = nt * 16 + nn;
        float bv = BIAS ? bias[col] : 0.0f;
        #pragma unroll
        for (int i = 0; i < 8; ++i) {
            long r = row0 + i + 8 * half;
            float v = acc[nt][i] + bv;
            if (ADDX) v += X[r * (long)ND + col];
            if (RELU) v = fmaxf(v, 0.0f);
            C[r * (long)ND + col] = v;
        }
    }
}

// ---------------- fused CFConv filter + gather + scatter-add ----------------
// Per 16-edge tile (one wave):
//   ea = exp(GCOEF*(ew-off_g)^2) computed in-register as WMMA A-fragments (K padded 50->64)
//   mid = relu(ea @ w1 + b1)  -> per-wave LDS (f16, padded pitch)
//   Wf  = (mid @ w2 + b2) * cwin
//   agg[col[e],:] += h[row[e],:] * Wf[e,:]   (global f32 atomics, L2-resident)
__global__ void __launch_bounds__(256)
filter_agg_kernel(const float* __restrict__ ew, const float* __restrict__ cwin,
                  const int* __restrict__ erow, const int* __restrict__ ecol,
                  const float* __restrict__ w1, const float* __restrict__ b1,
                  const float* __restrict__ w2, const float* __restrict__ b2,
                  const float* __restrict__ h, float* __restrict__ agg) {
    constexpr int KP = 72;                       // padded pitch for K=64
    __shared__ __align__(16) _Float16 sW1[64 * KP];   // K-major, zero-padded g>=50
    __shared__ __align__(16) _Float16 sW2[64 * KP];   // K-major
    __shared__ __align__(16) _Float16 smid[8][16 * KP];
    const int tid = threadIdx.x;
    const int wv = tid >> 5, lane = tid & 31;
    const int half = lane >> 4, nn = lane & 15;
    const long tile0 = ((long)blockIdx.x * 8 + wv) * 16;

    // stage weights once per block (coalesced over n; transpose into K-major)
    for (int idx = tid; idx < 64 * 64; idx += 256) {
        int k = idx >> 6, n = idx & 63;
        sW1[n * KP + k] = (k < G) ? (_Float16)w1[k * F + n] : (_Float16)0.0f;
        sW2[n * KP + k] = (_Float16)w2[idx];
    }
    const float ewm = ew[tile0 + nn];   // distance for this lane's A-row edge
    __syncthreads();

    // GEMM1: [16 x 64pad] ea  @  [64pad x 64] w1   (A computed analytically, branchless)
    v8f acc[4] = {};
    #pragma unroll
    for (int kt = 0; kt < 2; ++kt) {
        v16h a{};
        #pragma unroll
        for (int j = 0; j < 16; ++j) {
            int g = kt * 32 + ((j >> 3) << 4) + (half << 3) + (j & 7);  // two 8-runs
            float d = ewm - (float)g * GSTEP;
            float val = __expf(GCOEF * d * d);
            a[j] = (_Float16)((g < G) ? val : 0.0f);
        }
        #pragma unroll
        for (int nt = 0; nt < 4; ++nt) {
            v16h b = frag_b_from_lds(sW1 + (nt * 16 + nn) * KP + kt * 32 + half * 8);
            acc[nt] = __builtin_amdgcn_wmma_f32_16x16x32_f16(
                false, a, false, b, (short)0, acc[nt], false, false);
        }
    }
    // relu(. + b1) -> per-wave LDS tile (f16)
    _Float16* mid = smid[wv];
    #pragma unroll
    for (int nt = 0; nt < 4; ++nt) {
        int col = nt * 16 + nn;
        float bv = b1[col];
        #pragma unroll
        for (int i = 0; i < 8; ++i)
            mid[(i + 8 * half) * KP + col] = (_Float16)fmaxf(acc[nt][i] + bv, 0.0f);
    }
    __syncthreads();

    // GEMM2: [16 x 64] mid @ [64 x 64] w2
    v8f acc2[4] = {};
    #pragma unroll
    for (int kt = 0; kt < 2; ++kt) {
        v16h a = frag_a_from_lds(mid + nn * KP + kt * 32 + half * 8);
        #pragma unroll
        for (int nt = 0; nt < 4; ++nt) {
            v16h b = frag_b_from_lds(sW2 + (nt * 16 + nn) * KP + kt * 32 + half * 8);
            acc2[nt] = __builtin_amdgcn_wmma_f32_16x16x32_f16(
                false, a, false, b, (short)0, acc2[nt], false, false);
        }
    }
    // epilogue: Wf = (acc2+b2)*cwin ; agg[col] += h[row]*Wf
    #pragma unroll
    for (int i = 0; i < 8; ++i) {
        long e = tile0 + i + 8 * half;
        int r = erow[e], c = ecol[e];
        float cw = cwin[e];
        #pragma unroll
        for (int nt = 0; nt < 4; ++nt) {
            int f = nt * 16 + nn;
            float wf = (acc2[nt][i] + b2[f]) * cw;
            atomicAdd(&agg[(size_t)c * F + f], h[(size_t)r * F + f] * wf);
        }
    }
}

// ---------------- final readout: relu(pool2 @ out_w1 + b1) @ out_w2 + b2 ----------------
__global__ void out_kernel(const float* __restrict__ p2, const float* __restrict__ w1,
                           const float* __restrict__ b1, const float* __restrict__ w2,
                           const float* __restrict__ b2, float* __restrict__ out) {
    __shared__ float row[H];
    __shared__ float part[H];
    int m = blockIdx.x, j = threadIdx.x;  // blockDim = 128
    row[j] = p2[(size_t)m * H + j];
    __syncthreads();
    float t = 0.0f;
    for (int k = 0; k < H; ++k) t += row[k] * w1[k * H + j];
    t = fmaxf(t + b1[j], 0.0f) * w2[j];
    part[j] = t;
    __syncthreads();
    for (int o = 64; o > 0; o >>= 1) {
        if (j < o) part[j] += part[j + o];
        __syncthreads();
    }
    if (j == 0) out[m] = part[0] + b2[0];
}

// ---------------- host orchestration ----------------
extern "C" void kernel_launch(void* const* d_in, const int* in_sizes, int n_in,
                              void* d_out, int out_size, void* d_ws, size_t ws_size,
                              hipStream_t stream) {
    (void)in_sizes; (void)n_in; (void)out_size; (void)ws_size;

    const int*   x_topo    = (const int*)d_in[0];
    const float* pos       = (const float*)d_in[1];
    const int*   eic       = (const int*)d_in[2];
    const int*   eig       = (const int*)d_in[3];
    const int*   eattr     = (const int*)d_in[4];
    const int*   cnb       = (const int*)d_in[5];
    const int*   posb      = (const int*)d_in[6];
    const int*   confb     = (const int*)d_in[7];
    const float* atom_emb  = (const float*)d_in[8];
    const float* cf_lin1_w = (const float*)d_in[9];
    const float* cf_mlp_w1 = (const float*)d_in[10];
    const float* cf_mlp_b1 = (const float*)d_in[11];
    const float* cf_mlp_w2 = (const float*)d_in[12];
    const float* cf_mlp_b2 = (const float*)d_in[13];
    const float* cf_lin2_w = (const float*)d_in[14];
    const float* cf_lin2_b = (const float*)d_in[15];
    const float* lin_w     = (const float*)d_in[16];
    const float* lin_b     = (const float*)d_in[17];
    const float* bond_emb  = (const float*)d_in[18];
    const float* gin_w1    = (const float*)d_in[19];
    const float* gin_b1    = (const float*)d_in[20];
    const float* gin_w2    = (const float*)d_in[21];
    const float* gin_b2    = (const float*)d_in[22];
    const float* gin_bn_g  = (const float*)d_in[23];
    const float* gin_bn_b  = (const float*)d_in[24];
    const float* bn_g      = (const float*)d_in[25];
    const float* bn_b      = (const float*)d_in[26];
    const float* gin_eps   = (const float*)d_in[27];
    const float* out_w1    = (const float*)d_in[28];
    const float* out_b1    = (const float*)d_in[29];
    const float* out_w2    = (const float*)d_in[30];
    const float* out_b2    = (const float*)d_in[31];

    float* w = (float*)d_ws;
    float* ew    = w; w += (size_t)E_CONF;
    float* cwin  = w; w += (size_t)E_CONF;
    float* xbuf  = w; w += (size_t)N_NODE * H;
    float* cfout = w; w += (size_t)N_NODE * H;
    float* hbuf  = w; w += (size_t)N_NODE * F;
    float* agg   = w; w += (size_t)N_NODE * F;
    float* xagg  = w; w += (size_t)N_TOPO * H;
    float* magg  = w; w += (size_t)N_TOPO * H;
    float* y1    = w; w += (size_t)N_TOPO * H;
    float* y2    = w; w += (size_t)N_TOPO * H;
    float* bnsc  = w; w += H;
    float* bnsh  = w; w += H;
    float* pool1 = w; w += (size_t)NCONF * H;
    float* pool2 = w; w += (size_t)M_MOL * H;

    const int* erow = eic;
    const int* ecol = eic + E_CONF;
    const int* grow = eig;
    const int* gcol = eig + E_GRAPH;

    auto cdiv = [](size_t a, size_t b) { return (unsigned)((a + b - 1) / b); };

    // precompute edge geometry + node embeddings
    edge_kernel<<<cdiv(E_CONF, 256), 256, 0, stream>>>(pos, erow, ecol, ew, cwin);
    embed_kernel<<<cdiv((size_t)N_NODE * H, 256), 256, 0, stream>>>(atom_emb, x_topo, cnb, xbuf);

    for (int l = 0; l < L; ++l) {
        // ---- GIN aggregation of current x (before x is replaced) ----
        fill_kernel<<<cdiv((size_t)N_TOPO * H, 256), 256, 0, stream>>>(xagg, -INFINITY, (size_t)N_TOPO * H);
        smax_kernel<<<cdiv((size_t)N_NODE * H, 256), 256, 0, stream>>>(xbuf, cnb, xagg, N_NODE);

        // ---- CFConv branch ----
        gemm_kernel<128, 64, false, false, false><<<N_NODE / 64, 128, 0, stream>>>(
            xbuf, cf_lin1_w + (size_t)l * H * F, nullptr, nullptr, hbuf);
        fill_kernel<<<cdiv((size_t)N_NODE * F, 256), 256, 0, stream>>>(agg, 0.0f, (size_t)N_NODE * F);
        filter_agg_kernel<<<E_CONF / 128, 256, 0, stream>>>(
            ew, cwin, erow, ecol,
            cf_mlp_w1 + (size_t)l * G * F, cf_mlp_b1 + (size_t)l * F,
            cf_mlp_w2 + (size_t)l * F * F, cf_mlp_b2 + (size_t)l * F,
            hbuf, agg);
        gemm_kernel<64, 128, true, true, false><<<N_NODE / 64, 128, 0, stream>>>(
            agg, cf_lin2_w + (size_t)l * F * H, cf_lin2_b + (size_t)l * H, nullptr, cfout);
        // in-place: cfout = cfout @ lin_w + lin_b + x  (tile staged to LDS before writes)
        gemm_kernel<128, 128, true, false, true><<<N_NODE / 64, 128, 0, stream>>>(
            cfout, lin_w + (size_t)l * H * H, lin_b + (size_t)l * H, xbuf, cfout);

        // ---- GIN branch on topo graph ----
        fill_kernel<<<cdiv((size_t)N_TOPO * H, 256), 256, 0, stream>>>(magg, 0.0f, (size_t)N_TOPO * H);
        gin_msg_kernel<<<cdiv((size_t)E_GRAPH * H, 256), 256, 0, stream>>>(
            xagg, bond_emb + (size_t)l * BOND_TYPES * H, grow, gcol, eattr, magg);
        gin_pre_kernel<<<cdiv((size_t)N_TOPO * H, 256), 256, 0, stream>>>(xagg, magg, gin_eps, l);
        gemm_kernel<128, 128, true, false, false><<<N_TOPO / 64, 128, 0, stream>>>(
            magg, gin_w1 + (size_t)l * H * H, gin_b1 + (size_t)l * H, nullptr, y1);
        bn_stats_kernel<<<H, 256, 0, stream>>>(y1, N_TOPO, gin_bn_g + (size_t)l * H, gin_bn_b + (size_t)l * H, bnsc, bnsh);
        bn_apply_kernel<<<cdiv((size_t)N_TOPO * H, 256), 256, 0, stream>>>(y1, bnsc, bnsh, 1);
        gemm_kernel<128, 128, true, false, false><<<N_TOPO / 64, 128, 0, stream>>>(
            y1, gin_w2 + (size_t)l * H * H, gin_b2 + (size_t)l * H, nullptr, y2);
        bn_stats_kernel<<<H, 256, 0, stream>>>(y2, N_TOPO, bn_g + (size_t)l * H, bn_b + (size_t)l * H, bnsc, bnsh);
        bn_apply_kernel<<<cdiv((size_t)N_TOPO * H, 256), 256, 0, stream>>>(y2, bnsc, bnsh, 0);

        // ---- combine: x = relu(cfout + hg[cnb]) ----
        combine_kernel<<<cdiv((size_t)N_NODE * H, 256), 256, 0, stream>>>(cfout, y2, cnb, xbuf);
    }

    // ---- pooling + readout ----
    fill_kernel<<<cdiv((size_t)NCONF * H, 256), 256, 0, stream>>>(pool1, -INFINITY, (size_t)NCONF * H);
    smax_kernel<<<cdiv((size_t)N_NODE * H, 256), 256, 0, stream>>>(xbuf, posb, pool1, N_NODE);
    fill_kernel<<<cdiv((size_t)M_MOL * H, 256), 256, 0, stream>>>(pool2, -INFINITY, (size_t)M_MOL * H);
    smax_kernel<<<cdiv((size_t)NCONF * H, 256), 256, 0, stream>>>(pool1, confb, pool2, NCONF);
    out_kernel<<<M_MOL, 128, 0, stream>>>(pool2, out_w1, out_b1, out_w2, out_b2, (float*)d_out);
}
